// VDSelfAttention_6021544149237
// MI455X (gfx1250) — matrix operand order repaired and checked by reading, not compiled, not run on hardware
//
#include <hip/hip_runtime.h>

// MI455X (gfx1250, wave32) fused self-attention:
//   K1: Q/K/V = X @ W^T + b   via v_wmma_f32_16x16x32_bf16 (64-deep K steps)
//   K2: flash attention, 64-wide KV chunks, async global->LDS for V tiles
//
// LDS fragment layouts (A reads are 16B-aligned b128 loads, B reads are per element):
//   A-frag: lane = M row, elements = K with half-swizzle (pairs contiguous)
//   B-frag: lane = N col, elements = K  -> stage W and K *transposed* in LDS
//   C/D:    reg j -> row (j | j+8 by lane half), lanes = N
// Transposed staging writes packed pairs (-> ds_store_2addr_b32),
// softmax uses native v_exp_f32 via __builtin_amdgcn_exp2f.

typedef __attribute__((ext_vector_type(16))) __bf16 v16bf;
typedef __attribute__((ext_vector_type(8)))  __bf16 v8bf;
typedef __attribute__((ext_vector_type(2)))  __bf16 v2bf;
typedef __attribute__((ext_vector_type(8)))  float  v8f;

#define B_  2
#define S_  2048
#define D_  1024
#define H_  16
#define DH_ 64

#define LOG2E 1.4426950408889634f

static __device__ inline v16bf cat8(v8bf lo, v8bf hi) {
  return __builtin_shufflevector(lo, hi, 0, 1, 2, 3, 4, 5, 6, 7,
                                 8, 9, 10, 11, 12, 13, 14, 15);
}

static __device__ inline v8f wmma_bf16(v16bf a, v16bf b, v8f c) {
  return __builtin_amdgcn_wmma_f32_16x16x32_bf16(false, a, false, b, (short)0,
                                                 c, false, false);
}

static __device__ inline float fast_exp2(float x) {
  return __builtin_amdgcn_exp2f(x);   // bare v_exp_f32 (native 2^x)
}

// ---------------------------------------------------------------------------
// Kernel 1: out[m,n] = sum_k X[m,k]*W[n,k] + bias[n] -> bf16 [B][H][S][64]
// Block = 64x64 tile, 8 waves, z = {q,k,v}. K stepped by 64.
// ---------------------------------------------------------------------------
__global__ __launch_bounds__(256)
void qkv_proj_kernel(const float* __restrict__ xq, const float* __restrict__ xk,
                     const float* __restrict__ xv,
                     const float* __restrict__ Wq, const float* __restrict__ Wk,
                     const float* __restrict__ Wv,
                     const float* __restrict__ bq, const float* __restrict__ bk,
                     const float* __restrict__ bv,
                     __bf16* __restrict__ qo, __bf16* __restrict__ ko,
                     __bf16* __restrict__ vo) {
  const int sel = blockIdx.z;
  const float* __restrict__ X    = (sel == 0) ? xq : (sel == 1) ? xk : xv;
  const float* __restrict__ W    = (sel == 0) ? Wq : (sel == 1) ? Wk : Wv;
  const float* __restrict__ bias = (sel == 0) ? bq : (sel == 1) ? bk : bv;
  __bf16* __restrict__ O         = (sel == 0) ? qo : (sel == 1) ? ko : vo;

  __shared__ __align__(16) __bf16 As[64][72];    // M rows x 64 K (144B rows)
  __shared__ __align__(16) __bf16 WsT[64][72];   // K rows x 64 N (transposed W)

  const int tid  = threadIdx.x;
  const int lane = tid & 31;
  const int wave = tid >> 5;
  const int m0 = blockIdx.y * 64;
  const int n0 = blockIdx.x * 64;
  const int wm = (wave & 3) * 16;                // wave M sub-tile
  const int wn = (wave >> 2) * 32;               // wave N sub-tile (2x16)

  const int l  = lane & 15;
  const int kh = (lane >> 4) * 8;                // A-frag K half-swizzle

  v8f c0 = {}; v8f c1 = {};

  const int cr = tid >> 2;                       // X staging row 0..63
  const int cc = (tid & 3) * 16;                 // X staging col 0,16,32,48
  const int ra = tid & 31;                       // W staging: n-pair 0..31
  const int dg = (tid >> 5) * 8;                 // W staging: k cols dg..dg+7

  for (int k0 = 0; k0 < D_; k0 += 64) {
    // ---- stage X (row-major bf16, 2 x b128 LDS stores) ----
    {
      const float4* xp = (const float4*)(X + (size_t)(m0 + cr) * D_ + k0 + cc);
      float4 x0 = xp[0], x1 = xp[1], x2 = xp[2], x3 = xp[3];
      v8bf e0, e1;
      e0[0] = (__bf16)x0.x; e0[1] = (__bf16)x0.y; e0[2] = (__bf16)x0.z; e0[3] = (__bf16)x0.w;
      e0[4] = (__bf16)x1.x; e0[5] = (__bf16)x1.y; e0[6] = (__bf16)x1.z; e0[7] = (__bf16)x1.w;
      e1[0] = (__bf16)x2.x; e1[1] = (__bf16)x2.y; e1[2] = (__bf16)x2.z; e1[3] = (__bf16)x2.w;
      e1[4] = (__bf16)x3.x; e1[5] = (__bf16)x3.y; e1[6] = (__bf16)x3.z; e1[7] = (__bf16)x3.w;
      *(v8bf*)&As[cr][cc]     = e0;
      *(v8bf*)&As[cr][cc + 8] = e1;
    }
    // ---- stage W transposed: 8x2 patch -> packed pair stores ----
    {
      const float4* w0p = (const float4*)(W + (size_t)(n0 + 2 * ra) * D_ + k0 + dg);
      const float4* w1p = (const float4*)(W + (size_t)(n0 + 2 * ra + 1) * D_ + k0 + dg);
      float4 wa0 = w0p[0], wa1 = w0p[1];         // row n = 2ra,   k = dg..dg+7
      float4 wb0 = w1p[0], wb1 = w1p[1];         // row n = 2ra+1, k = dg..dg+7
      float ea[8] = {wa0.x, wa0.y, wa0.z, wa0.w, wa1.x, wa1.y, wa1.z, wa1.w};
      float eb[8] = {wb0.x, wb0.y, wb0.z, wb0.w, wb1.x, wb1.y, wb1.z, wb1.w};
#pragma unroll
      for (int i = 0; i < 8; ++i) {
        v2bf pr; pr[0] = (__bf16)ea[i]; pr[1] = (__bf16)eb[i];
        *(v2bf*)&WsT[dg + i][2 * ra] = pr;
      }
    }
    __syncthreads();

    // ---- fragments (A: b128 LDS loads, B: per-element loads) ----
    const __bf16* ap = &As[wm + l][0];
    v16bf a0 = cat8(*(const v8bf*)(ap + kh),      *(const v8bf*)(ap + 16 + kh));
    v16bf a1 = cat8(*(const v8bf*)(ap + 32 + kh), *(const v8bf*)(ap + 48 + kh));

    v16bf b00, b01, b10, b11;
#pragma unroll
    for (int s = 0; s < 16; ++s) {
      const int kk = kh + (s & 7) + ((s >> 3) << 4);
      b00[s] = WsT[kk][wn + l];      b01[s] = WsT[kk][wn + 16 + l];
      b10[s] = WsT[32 + kk][wn + l]; b11[s] = WsT[32 + kk][wn + 16 + l];
    }

    c0 = wmma_bf16(a0, b00, c0);
    c0 = wmma_bf16(a1, b10, c0);
    c1 = wmma_bf16(a0, b01, c1);
    c1 = wmma_bf16(a1, b11, c1);
    __syncthreads();
  }

  // ---- bias + store to per-head bf16 layout ----
  const int mh = (lane >> 4) * 8;
  const float bn0 = bias[n0 + wn + l];
  const float bn1 = bias[n0 + wn + 16 + l];
#pragma unroll
  for (int j = 0; j < 8; ++j) {
    int m = m0 + wm + mh + j;
    int bb = m >> 11, s = m & (S_ - 1);
    {
      int n = n0 + wn + l;
      int h = n >> 6, d = n & 63;
      O[((((size_t)bb * H_ + h) * S_ + s) << 6) + d] = (__bf16)(c0[j] + bn0);
    }
    {
      int n = n0 + wn + 16 + l;
      int h = n >> 6, d = n & 63;
      O[((((size_t)bb * H_ + h) * S_ + s) << 6) + d] = (__bf16)(c1[j] + bn1);
    }
  }
}

// ---------------------------------------------------------------------------
// Kernel 2: flash attention. Block = (b, h, 128 q-rows); 8 waves x 16 q-rows.
// KV in chunks of 64: K staged transposed (packed b32), V via async->LDS.
// Softmax in base-2 domain (native v_exp_f32).
// ---------------------------------------------------------------------------
__global__ __launch_bounds__(256)
void attn_kernel(const __bf16* __restrict__ Q, const __bf16* __restrict__ K,
                 const __bf16* __restrict__ V, const int* __restrict__ mask,
                 float* __restrict__ out) {
  const int tid  = threadIdx.x;
  const int lane = tid & 31;
  const int wave = tid >> 5;
  const int h = blockIdx.y;
  const int b = blockIdx.z;

  __shared__ __align__(16) __bf16 KtT[64][72];     // d rows x 64 kv (transposed)
  __shared__ __align__(16) __bf16 Vt[64][72];      // kv rows x 64 d
  __shared__ float Msk[64];
  __shared__ __align__(16) __bf16 Pt[8][16][72];   // per-wave 16x64 P tile

  const size_t head_base = ((size_t)b * H_ + h) * S_ * DH_;
  const int q0 = blockIdx.x * 128 + wave * 16;

  const int l  = lane & 15;
  const int kh = (lane >> 4) * 8;
  const int mh = (lane >> 4) * 8;

  // resident Q A-fragments (d 0..31, 32..63)
  v16bf qa0, qa1;
  {
    const __bf16* qp = Q + head_base + (size_t)(q0 + l) * DH_;
    qa0 = cat8(*(const v8bf*)(qp + kh),      *(const v8bf*)(qp + 16 + kh));
    qa1 = cat8(*(const v8bf*)(qp + 32 + kh), *(const v8bf*)(qp + 48 + kh));
  }

  float mrun[8], lrun[8];
  v8f acc[4] = {};
#pragma unroll
  for (int j = 0; j < 8; ++j) { mrun[j] = -3.0e38f; lrun[j] = 0.0f; }

  const int cr = tid >> 2;                 // V staging row 0..63
  const int cc = (tid & 3) * 16;           // V staging col 0,16,32,48
  const int ra = tid & 31;                 // K staging: kv-pair 0..31
  const int dg = (tid >> 5) * 8;           // K staging: d cols dg..dg+7

  const float SC = 0.125f * LOG2E;         // 1/sqrt(64) folded into base-2

  for (int kv0 = 0; kv0 < S_; kv0 += 64) {
    // ---- V: async global->LDS b128 copies (2 x 16B per thread) ----
    {
      unsigned lv = (unsigned)(unsigned long long)&Vt[cr][cc];
      unsigned long long gv =
          (unsigned long long)(V + head_base + (size_t)(kv0 + cr) * DH_ + cc);
      asm volatile("global_load_async_to_lds_b128 %0, %1, off"
                   :: "v"(lv), "v"(gv) : "memory");
      asm volatile("global_load_async_to_lds_b128 %0, %1, off"
                   :: "v"(lv + 16u), "v"(gv + 16ull) : "memory");
    }
    // ---- K: 8x2 patch, store transposed as packed pair stores ----
    {
      const uint4* k0p =
          (const uint4*)(K + head_base + (size_t)(kv0 + 2 * ra) * DH_ + dg);
      const uint4* k1p =
          (const uint4*)(K + head_base + (size_t)(kv0 + 2 * ra + 1) * DH_ + dg);
      uint4 ka = k0p[0], kb = k1p[0];
      const __bf16* e0 = (const __bf16*)&ka;   // kv = 2ra,   d = dg..dg+7
      const __bf16* e1 = (const __bf16*)&kb;   // kv = 2ra+1, d = dg..dg+7
#pragma unroll
      for (int i = 0; i < 8; ++i) {
        v2bf pr; pr[0] = e0[i]; pr[1] = e1[i];
        *(v2bf*)&KtT[dg + i][2 * ra] = pr;
      }
    }
    if (tid < 64)
      Msk[tid] = (1.0e30f * LOG2E) * (float)mask[b * S_ + kv0 + tid];
    if (kv0 + 64 < S_)   // prefetch next K chunk (-> global_prefetch_b8)
      __builtin_prefetch((const void*)(K + head_base +
                                       (size_t)(kv0 + 64) * DH_ + tid * 16), 0, 0);
    asm volatile("s_wait_asynccnt 0x0" ::: "memory");
    __syncthreads();

    // ---- scores: 4 n-tiles x 2 d-steps = 8 WMMAs ----
    v8f s[4];
#pragma unroll
    for (int t = 0; t < 4; ++t) {
      v8f z = {};
      v16bf blo, bhi;
#pragma unroll
      for (int s = 0; s < 16; ++s) {
        const int kk = kh + (s & 7) + ((s >> 3) << 4);
        blo[s] = KtT[kk][t * 16 + l]; bhi[s] = KtT[32 + kk][t * 16 + l];
      }
      z = wmma_bf16(qa0, blo, z);
      s[t] = wmma_bf16(qa1, bhi, z);
    }

    // ---- scale + mask + online softmax (base-2, row stats in lane halves) --
    float mk[4];
#pragma unroll
    for (int t = 0; t < 4; ++t) mk[t] = Msk[t * 16 + l];
#pragma unroll
    for (int j = 0; j < 8; ++j) {
#pragma unroll
      for (int t = 0; t < 4; ++t) s[t][j] = s[t][j] * SC - mk[t];
      float mx = fmaxf(fmaxf(s[0][j], s[1][j]), fmaxf(s[2][j], s[3][j]));
#pragma unroll
      for (int off = 8; off >= 1; off >>= 1) mx = fmaxf(mx, __shfl_xor(mx, off, 32));
      float mnew  = fmaxf(mrun[j], mx);
      float scale = fast_exp2(mrun[j] - mnew);
      float rs = 0.0f;
#pragma unroll
      for (int t = 0; t < 4; ++t) { s[t][j] = fast_exp2(s[t][j] - mnew); rs += s[t][j]; }
#pragma unroll
      for (int off = 8; off >= 1; off >>= 1) rs += __shfl_xor(rs, off, 32);
      lrun[j] = lrun[j] * scale + rs;
      mrun[j] = mnew;
#pragma unroll
      for (int t = 0; t < 4; ++t) acc[t][j] *= scale;
    }

    // ---- P: C layout -> A layout via per-wave LDS tile ----
#pragma unroll
    for (int j = 0; j < 8; ++j)
#pragma unroll
      for (int t = 0; t < 4; ++t)
        Pt[wave][mh + j][t * 16 + l] = (__bf16)s[t][j];
    __syncthreads();

    const __bf16* pp = &Pt[wave][l][0];
    v16bf pa0 = cat8(*(const v8bf*)(pp + kh),      *(const v8bf*)(pp + 16 + kh));
    v16bf pa1 = cat8(*(const v8bf*)(pp + 32 + kh), *(const v8bf*)(pp + 48 + kh));

    // ---- P @ V: 4 d-tiles x 2 kv-steps = 8 WMMAs ----
#pragma unroll
    for (int dt = 0; dt < 4; ++dt) {
      v16bf bvA, bvB;
#pragma unroll
      for (int s = 0; s < 16; ++s) {
        const int kk = kh + (s & 7) + ((s >> 3) << 4);
        bvA[s] = Vt[kk][dt * 16 + l]; bvB[s] = Vt[32 + kk][dt * 16 + l];
      }
      acc[dt] = wmma_bf16(pa0, bvA, acc[dt]);
      acc[dt] = wmma_bf16(pa1, bvB, acc[dt]);
    }
    __syncthreads();
  }

  // ---- normalize + store fp32, out[b, row, h*64 + d] ----
#pragma unroll
  for (int j = 0; j < 8; ++j) {
    float inv = 1.0f / lrun[j];
    int row = q0 + mh + j;
    float* op = out + ((size_t)b * S_ + row) * D_ + h * DH_;
    op[l]      = acc[0][j] * inv;
    op[16 + l] = acc[1][j] * inv;
    op[32 + l] = acc[2][j] * inv;
    op[48 + l] = acc[3][j] * inv;
  }
}

// ---------------------------------------------------------------------------
extern "C" void kernel_launch(void* const* d_in, const int* in_sizes, int n_in,
                              void* d_out, int out_size, void* d_ws, size_t ws_size,
                              hipStream_t stream) {
  const float* q_head = (const float*)d_in[0];
  const float* k_head = (const float*)d_in[1];
  const float* v_head = (const float*)d_in[2];
  const int*   mask   = (const int*)d_in[3];
  const float* Wq     = (const float*)d_in[4];
  const float* bq     = (const float*)d_in[5];
  const float* Wk     = (const float*)d_in[6];
  const float* bk     = (const float*)d_in[7];
  const float* Wv     = (const float*)d_in[8];
  const float* bv     = (const float*)d_in[9];
  float* out = (float*)d_out;

  const size_t per = (size_t)B_ * H_ * S_ * DH_;   // 4M bf16 elems = 8 MB each
  __bf16* qb = (__bf16*)d_ws;
  __bf16* kb = qb + per;
  __bf16* vb = kb + per;

  dim3 g1(D_ / 64, (B_ * S_) / 64, 3);             // 16 x 64 x 3
  qkv_proj_kernel<<<g1, 256, 0, stream>>>(q_head, k_head, v_head,
                                          Wq, Wk, Wv, bq, bk, bv, qb, kb, vb);

  dim3 g2(S_ / 128, H_, B_);                       // 16 x 16 x 2
  attn_kernel<<<g2, 256, 0, stream>>>(qb, kb, vb, mask, out);
}
